// drebin_42288247996442
// MI455X (gfx1250) — matrix operand order repaired and checked by reading, not compiled, not run on hardware
//
#include <hip/hip_runtime.h>
#include <hip/hip_bf16.h>

// ---------------------------------------------------------------------------
// 16-qubit state-vector simulator for MI455X (gfx1250, wave32).
//
// Design:
//  * State: 64 batches x 65536 complex64 (float2) = 32 MB in d_ws -> L2-resident.
//  * Bit convention: wire w <-> bit (15-w) of the flat state index
//    (matches state.reshape((2,)*16).reshape(-1) in the reference).
//  * Slice bit = bit 15 (wire 0). A 32768-amp slice = 256 KB fits in the
//    320 KB LDS of one WGP. Per layer:
//      k_local_A : fused Rot(w)@RX(x_w) for wires 1..15, LDS-resident
//      k_cross_1q: fused gate on wire 0 (pairs across bit 15), elementwise
//      k_local_B : CRX(0,1) (control = slice bit, uniform) + CRX(1..14),
//                  LDS-resident
//      k_cross_cx: CRX(15,0) (target = bit 15), elementwise, control = bit 0
//  * Readout: per-batch block reduction of probabilities -> z[64][16].
//  * FC + log_softmax: v_wmma_f32_16x16x32_f16 (4 waves, one 16x16 tile each),
//    K padded 16->32, N padded 10->16 with zeros.
// ---------------------------------------------------------------------------

#define NQ       16
#define NSTATES  65536      // 2^16
#define HALF_ST  32768      // 2^15 (one bit-15 slice)
#define B        64
#define NL       5
#define NC       10

struct C4 { float2 m00, m01, m10, m11; };

__device__ __forceinline__ float2 cmul(float2 a, float2 b) {
    return make_float2(a.x * b.x - a.y * b.y, a.x * b.y + a.y * b.x);
}
__device__ __forceinline__ float2 cadd(float2 a, float2 b) {
    return make_float2(a.x + b.x, a.y + b.y);
}
__device__ __forceinline__ float2 csr(float2 a, float s) {
    return make_float2(a.x * s, a.y * s);
}

// Fused U = Rot(phi,theta,omega) @ RX(xt)  (RX applied first, as in the ref)
__device__ __forceinline__ C4 fused_rot_rx(float xt, float phi, float th, float om) {
    float sx, cx; __sincosf(0.5f * xt, &sx, &cx);
    float st, ct; __sincosf(0.5f * th, &st, &ct);
    float ap = -0.5f * (phi + om), am = -0.5f * (phi - om);
    float sp, cp; __sincosf(ap, &sp, &cp);
    float sm_, cm_; __sincosf(am, &sm_, &cm_);
    float2 ep = make_float2(cp, sp);     // e^{-i(phi+om)/2}
    float2 em = make_float2(cm_, sm_);   // e^{-i(phi-om)/2}
    float2 R00 = csr(ep, ct);
    float2 R01 = make_float2(-em.x * st,  em.y * st);   // -conj(em)*st
    float2 R10 = csr(em, st);
    float2 R11 = make_float2( ep.x * ct, -ep.y * ct);   //  conj(ep)*ct
    float2 misx = make_float2(0.f, -sx);                // -i*sin(x/2)
    C4 A;
    A.m00 = cadd(csr(R00, cx), cmul(R01, misx));
    A.m01 = cadd(cmul(R00, misx), csr(R01, cx));
    A.m10 = cadd(csr(R10, cx), cmul(R11, misx));
    A.m11 = cadd(cmul(R10, misx), csr(R11, cx));
    return A;
}

__device__ __forceinline__ C4 rx4(float t) {
    float sx, cx; __sincosf(0.5f * t, &sx, &cx);
    C4 U;
    U.m00 = make_float2(cx, 0.f);  U.m11 = make_float2(cx, 0.f);
    U.m01 = make_float2(0.f, -sx); U.m10 = make_float2(0.f, -sx);
    return U;
}

__device__ __forceinline__ void bfly(float2& a0, float2& a1, const C4& U) {
    float2 n0 = cadd(cmul(U.m00, a0), cmul(U.m01, a1));
    float2 n1 = cadd(cmul(U.m10, a0), cmul(U.m11, a1));
    a0 = n0; a1 = n1;
}

// ---------------------------------------------------------------------------
// |0...0> initialization
__global__ void k_init(float2* state) {
    int gid = blockIdx.x * blockDim.x + threadIdx.x;   // 0 .. B*NSTATES-1
    state[gid] = ((gid & (NSTATES - 1)) == 0) ? make_float2(1.f, 0.f)
                                              : make_float2(0.f, 0.f);
}

// ---------------------------------------------------------------------------
// LDS-resident fused 1q gates on wires 1..15 (bits 14..0) for one slice.
__global__ void k_local_A(const float* __restrict__ x,
                          const float* __restrict__ rot,
                          float2* __restrict__ state, int n) {
    int b = blockIdx.x >> 1, h = blockIdx.x & 1;
    float2* g = state + (size_t)b * NSTATES + (size_t)h * HALF_ST;
    extern __shared__ char smem_raw[];
    float2* s = (float2*)smem_raw;

    for (int i = threadIdx.x; i < HALF_ST; i += blockDim.x) s[i] = g[i];
    __syncthreads();

    for (int w = 1; w < NQ; ++w) {
        float xt  = x[b * NQ + w];
        const float* r = rot + ((size_t)n * NQ + w) * 3;
        C4 U = fused_rot_rx(xt, r[0], r[1], r[2]);
        int t = 15 - w;                 // bit inside the slice (14..0)
        int mt = 1 << t;
        for (int i = threadIdx.x; i < (HALF_ST >> 1); i += blockDim.x) {
            int lo = i & (mt - 1);
            int i0 = ((i >> t) << (t + 1)) | lo;
            int i1 = i0 | mt;
            float2 a0 = s[i0], a1 = s[i1];
            bfly(a0, a1, U);
            s[i0] = a0; s[i1] = a1;
        }
        __syncthreads();
    }
    for (int i = threadIdx.x; i < HALF_ST; i += blockDim.x) g[i] = s[i];
}

// ---------------------------------------------------------------------------
// Fused 1q gate on wire 0 (bit 15): pairs (p, p+32768), elementwise.
__global__ void k_cross_1q(const float* __restrict__ x,
                           const float* __restrict__ rot,
                           float2* __restrict__ state, int n) {
    int gid = blockIdx.x * blockDim.x + threadIdx.x;   // B * HALF_ST threads
    int b = gid >> 15, p = gid & (HALF_ST - 1);
    const float* r = rot + ((size_t)n * NQ + 0) * 3;
    C4 U = fused_rot_rx(x[b * NQ + 0], r[0], r[1], r[2]);
    float2* g = state + (size_t)b * NSTATES;
    float2 a0 = g[p], a1 = g[p + HALF_ST];
    bfly(a0, a1, U);
    g[p] = a0; g[p + HALF_ST] = a1;
}

// ---------------------------------------------------------------------------
// LDS-resident CRX chain: CRX(0,1) (control = slice bit h), CRX(1..14).
__global__ void k_local_B(const float* __restrict__ crx,
                          float2* __restrict__ state, int n) {
    int b = blockIdx.x >> 1, h = blockIdx.x & 1;
    float2* g = state + (size_t)b * NSTATES + (size_t)h * HALF_ST;
    extern __shared__ char smem_raw[];
    float2* s = (float2*)smem_raw;

    for (int i = threadIdx.x; i < HALF_ST; i += blockDim.x) s[i] = g[i];
    __syncthreads();

    // CRX(0,1): control wire 0 = bit 15 = h (uniform). Target wire 1 = bit 14.
    {
        C4 U = rx4(crx[n * NQ + 0]);
        if (h == 1) {
            const int t = 14, mt = 1 << t;
            for (int i = threadIdx.x; i < (HALF_ST >> 1); i += blockDim.x) {
                int lo = i & (mt - 1);
                int i0 = ((i >> t) << (t + 1)) | lo;
                int i1 = i0 | mt;
                float2 a0 = s[i0], a1 = s[i1];
                bfly(a0, a1, U);
                s[i0] = a0; s[i1] = a1;
            }
        }
        __syncthreads();
    }

    // CRX(w, w+1), w = 1..14: control bit = 15-w, target bit = 14-w (adjacent,
    // control is the higher bit). Enumerate the 8192 control==1 target pairs.
    for (int w = 1; w <= 14; ++w) {
        C4 U = rx4(crx[n * NQ + w]);
        int t = 14 - w;               // target bit (13..0)
        int mt = 1 << t;
        for (int i = threadIdx.x; i < (HALF_ST >> 2); i += blockDim.x) {
            int lo = i & (mt - 1);
            int hi = i >> t;
            int i0 = (hi << (t + 2)) | (mt << 1) | lo;   // control bit set
            int i1 = i0 | mt;
            float2 a0 = s[i0], a1 = s[i1];
            bfly(a0, a1, U);
            s[i0] = a0; s[i1] = a1;
        }
        __syncthreads();
    }
    for (int i = threadIdx.x; i < HALF_ST; i += blockDim.x) g[i] = s[i];
}

// ---------------------------------------------------------------------------
// CRX(15,0): control = bit 0 (element-local), target = bit 15 (cross).
// Only odd p (control==1) pairs are touched.
__global__ void k_cross_cx(const float* __restrict__ crx,
                           float2* __restrict__ state, int n) {
    int gid = blockIdx.x * blockDim.x + threadIdx.x;   // B * 16384 threads
    int b = gid >> 14, q = gid & 16383;
    int p = 2 * q + 1;
    C4 U = rx4(crx[n * NQ + 15]);
    float2* g = state + (size_t)b * NSTATES;
    float2 a0 = g[p], a1 = g[p + HALF_ST];
    bfly(a0, a1, U);
    g[p] = a0; g[p + HALF_ST] = a1;
}

// ---------------------------------------------------------------------------
// <Z_w> readout: z_w = T - 2 * sum_{bit(15-w)==1} p(s), T = total prob.
__global__ void k_readout(const float2* __restrict__ state,
                          float* __restrict__ zbuf) {
    int b = blockIdx.x;
    const float2* g = state + (size_t)b * NSTATES;
    float acc[17];
#pragma unroll
    for (int i = 0; i < 17; ++i) acc[i] = 0.f;
    for (int idx = threadIdx.x; idx < NSTATES; idx += blockDim.x) {
        float2 a = g[idx];
        float p = a.x * a.x + a.y * a.y;
        acc[16] += p;
#pragma unroll
        for (int w = 0; w < NQ; ++w)
            acc[w] += p * (float)((idx >> (15 - w)) & 1);
    }
    __shared__ float sm[256];
    __shared__ float sres[17];
    for (int comp = 0; comp < 17; ++comp) {
        sm[threadIdx.x] = acc[comp];
        __syncthreads();
        for (int off = 128; off > 0; off >>= 1) {
            if ((int)threadIdx.x < off) sm[threadIdx.x] += sm[threadIdx.x + off];
            __syncthreads();
        }
        if (threadIdx.x == 0) sres[comp] = sm[0];
        __syncthreads();
    }
    if (threadIdx.x < NQ)
        zbuf[b * NQ + threadIdx.x] = sres[16] - 2.f * sres[threadIdx.x];
}

// ---------------------------------------------------------------------------
// FC (z @ W^T + b) via v_wmma_f32_16x16x32_f16, then per-row log_softmax.
// One block of 128 threads = 4 waves; wave v computes batch rows 16v..16v+15.
typedef __attribute__((ext_vector_type(16))) _Float16 v16h;
typedef __attribute__((ext_vector_type(8)))  float    v8f;

__global__ void k_fc(const float* __restrict__ z,
                     const float* __restrict__ fc_w,
                     const float* __restrict__ fc_b,
                     float* __restrict__ out) {
    __shared__ float lg[B][16];
    int lane = threadIdx.x & 31;
    int wv   = threadIdx.x >> 5;          // 0..3

    v16h a, bm;
#pragma unroll
    for (int e = 0; e < 16; ++e) {
        // A (16x32 f16, M x K): lane L holds row m=L&15; element e -> K index:
        //   k = (vgpr&3)*2 + (e&1) + (vgpr>>2)*16 + (L>>4)*8, vgpr = e>>1
        int j = e >> 1;
        int k = ((j & 3) * 2) + (e & 1) + ((j >> 2) * 16) + ((lane >> 4) * 8);
        int m = lane & 15;
        a[e] = (k < NQ) ? (_Float16)z[(wv * 16 + m) * NQ + k] : (_Float16)0.f;
        // B (32x16 f16, K x N): same K order as A;
        // column n = L&15, element e -> K = (e&7) + (e>>3)*16 + (L>>4)*8
        int kb = (e & 7) + ((e >> 3) * 16) + ((lane >> 4) * 8);
        int nn = lane & 15;
        bm[e] = (kb < NQ && nn < NC) ? (_Float16)fc_w[nn * NQ + kb]
                                     : (_Float16)0.f;
    }
    v8f c = {};
    c = __builtin_amdgcn_wmma_f32_16x16x32_f16(
            /*neg_a=*/false, a, /*neg_b=*/false, bm,
            /*c_mod=*/(short)0, c, /*reuse_a=*/false, /*reuse_b=*/false);

    // D layout: VGPR r -> lanes 0-15: (M=r, N=lane); lanes 16-31: (M=r+8, N=lane-16)
#pragma unroll
    for (int r = 0; r < 8; ++r) {
        int row = wv * 16 + ((lane >> 4) * 8) + r;
        int col = lane & 15;
        lg[row][col] = c[r];
    }
    __syncthreads();

    int t = threadIdx.x;
    if (t < B) {
        float l[NC];
        float mx = -3.0e38f;
        for (int nn = 0; nn < NC; ++nn) {
            l[nn] = lg[t][nn] + fc_b[nn];
            mx = fmaxf(mx, l[nn]);
        }
        float ssum = 0.f;
        for (int nn = 0; nn < NC; ++nn) ssum += expf(l[nn] - mx);
        float ls = mx + logf(ssum);
        for (int nn = 0; nn < NC; ++nn) out[t * NC + nn] = l[nn] - ls;
    }
}

// ---------------------------------------------------------------------------
extern "C" void kernel_launch(void* const* d_in, const int* in_sizes, int n_in,
                              void* d_out, int out_size, void* d_ws, size_t ws_size,
                              hipStream_t stream) {
    const float* x    = (const float*)d_in[0];   // (64,16)
    const float* rot  = (const float*)d_in[1];   // (5,16,3)
    const float* crx  = (const float*)d_in[2];   // (5,16,1)
    const float* fc_w = (const float*)d_in[3];   // (10,16)
    const float* fc_b = (const float*)d_in[4];   // (10,)
    float* out = (float*)d_out;                  // (64,10) log_softmax

    // Workspace: 32 MB state + 4 KB z buffer (requires ws_size >= ~33.6 MB)
    float2* state = (float2*)d_ws;
    float*  zbuf  = (float*)((char*)d_ws + (size_t)B * NSTATES * sizeof(float2));

    const size_t lds_bytes = (size_t)HALF_ST * sizeof(float2);   // 256 KB < 320 KB

    k_init<<<dim3((B * NSTATES) / 256), dim3(256), 0, stream>>>(state);

    for (int n = 0; n < NL; ++n) {
        k_local_A<<<dim3(B * 2), dim3(1024), lds_bytes, stream>>>(x, rot, state, n);
        k_cross_1q<<<dim3((B * HALF_ST) / 256), dim3(256), 0, stream>>>(x, rot, state, n);
        k_local_B<<<dim3(B * 2), dim3(1024), lds_bytes, stream>>>(crx, state, n);
        k_cross_cx<<<dim3((B * 16384) / 256), dim3(256), 0, stream>>>(crx, state, n);
    }

    k_readout<<<dim3(B), dim3(256), 0, stream>>>(state, zbuf);
    k_fc<<<dim3(1), dim3(128), 0, stream>>>(zbuf, fc_w, fc_b, out);
}
